// RelativeMultiheadAttention_704374637238
// MI455X (gfx1250) — matrix operand order repaired and checked by reading, not compiled, not run on hardware
//
#include <hip/hip_runtime.h>

// Workspace layout (~101 MB assumed <= ws_size):
//   5x WT bf16 512x512 | qu/qv/kk bf16 [B*H,S,64] | vt bf16 [B*H,64,S]
//   pp bf16 [B*H,2048,64] (row 2047 zeroed) | ctx bf16 [B*S,512]
//   xq/xk/xv bf16 [B*S,512] | xp bf16 [B*2047,512]

#define TPB 128

typedef __attribute__((ext_vector_type(16))) __bf16 v16bf;
typedef __attribute__((ext_vector_type(8)))  float  v8f;

struct U16B { unsigned int x, y, z, w; };  // trivially-copyable 16B chunk
union BFrag { v16bf v; unsigned short u[16]; U16B q[2]; };

__device__ __forceinline__ unsigned short f2bf(float f) {
  union { float f; unsigned int u; } x; x.f = f;
  return (unsigned short)((x.u + 0x7FFFu + ((x.u >> 16) & 1u)) >> 16);
}
__device__ __forceinline__ float bf2f(unsigned short h) {
  union { unsigned int u; float f; } x; x.u = ((unsigned int)h) << 16;
  return x.f;
}

// A-operand 16x32 bf16: lane half0 holds K{0..7,16..23}, half1 holds K{8..15,24..31}
__device__ __forceinline__ v16bf loadA_bf16(const unsigned short* row, int kc, int half) {
  BFrag f;
  f.q[0] = *(const U16B*)(row + kc * 32 + half * 8);
  f.q[1] = *(const U16B*)(row + kc * 32 + 16 + half * 8);
  return f.v;
}
// B-operand 32x16 bf16 (column data contiguous per lane): half0 K{0..7,16..23}, half1 K{8..15,24..31}
__device__ __forceinline__ v16bf loadB_bf16(const unsigned short* row, int kc, int half) {
  const unsigned short* p = row + kc * 32 + half * 8;
  BFrag f;
  f.q[0] = *(const U16B*)p;
  f.q[1] = *(const U16B*)(p + 16);
  return f.v;
}
__device__ __forceinline__ v8f wmma_bf16(v16bf a, v16bf b, v8f c) {
  return __builtin_amdgcn_wmma_f32_16x16x32_bf16(false, a, false, b, (short)0, c, false, false);
}
#define V8F_ZERO {0.f, 0.f, 0.f, 0.f, 0.f, 0.f, 0.f, 0.f}

// ---------------- streaming fp32 -> bf16 conversion (8 elems/thread) ----------
__global__ void cvt_bf16_kernel(const float* __restrict__ src,
                                unsigned short* __restrict__ dst, int n8) {
  int i = blockIdx.x * blockDim.x + threadIdx.x;
  if (i >= n8) return;
  const float* s = src + (long long)i * 8;
  BFrag f;
#pragma unroll
  for (int j = 0; j < 8; ++j) f.u[j] = f2bf(s[j]);
  *(U16B*)(dst + (long long)i * 8) = f.q[0];
}

// ---------------- prep: transpose+bf16 weights, zero p pad row ----------------
__global__ void prep_kernel(const float* __restrict__ Wq, const float* __restrict__ Wk,
                            const float* __restrict__ Wv, const float* __restrict__ Wp,
                            const float* __restrict__ Wo,
                            unsigned short* wqT, unsigned short* wkT, unsigned short* wvT,
                            unsigned short* wpT, unsigned short* woT,
                            unsigned short* pp) {
  int idx = blockIdx.x * blockDim.x + threadIdx.x;
  const int total = 5 * 512 * 512;
  if (idx < total) {
    int m = idx >> 18;
    int e = idx & 262143;
    int n = e >> 9, k = e & 511;
    const float* W = (m == 0) ? Wq : (m == 1) ? Wk : (m == 2) ? Wv : (m == 3) ? Wp : Wo;
    unsigned short* WT = (m == 0) ? wqT : (m == 1) ? wkT : (m == 2) ? wvT : (m == 3) ? wpT : woT;
    WT[n * 512 + k] = f2bf(W[k * 512 + n]);
  } else {
    int pidx = idx - total;
    if (pidx < 64 * 64) {
      int bh = pidx >> 6, d = pidx & 63;
      pp[((long long)bh * 2048 + 2047) * 64 + d] = 0;
    }
  }
}

// ------ generic WMMA GEMM: wave computes 16x64 strip, 4 independent accums ----
// MODE 0: qu/qv (+u_bias/+v_bias) -> [B*H,S,64]   MODE 1: k -> [B*H,S,64]
// MODE 2: vT -> [B*H,64,S]   MODE 3: p -> [B*H,2048,64] (rows<2047)
// MODE 4: fp32 out -> [rows,512]
template <int MODE>
__global__ void proj_gemm(const unsigned short* __restrict__ X,
                          const unsigned short* __restrict__ WT,
                          void* __restrict__ out0, void* __restrict__ out1,
                          const float* __restrict__ bias0, const float* __restrict__ bias1,
                          int rowsPerB, int nB, int rowTiles) {
  const int wave = threadIdx.x >> 5;
  const int lane = threadIdx.x & 31;
  const int tile = blockIdx.x * (TPB / 32) + wave;
  const int tilesPerB = rowTiles * 8;  // 8 column groups of 64
  const int bb = tile / tilesPerB;
  if (bb >= nB) return;
  const int rem = tile - bb * tilesPerB;
  const int rt = rem >> 3;
  const int cg = rem & 7;
  const int srow = lane & 15, half = lane >> 4;
  int rA = rt * 16 + srow;
  if (rA > rowsPerB - 1) rA = rowsPerB - 1;  // clamp (p: 2047 rows/b)
  const unsigned short* xrow = X + ((long long)bb * rowsPerB + rA) * 512;
  const unsigned short* w0 = WT + (cg * 64 + srow) * 512;
  const unsigned short* w1 = w0 + 16 * 512;
  const unsigned short* w2 = w0 + 32 * 512;
  const unsigned short* w3 = w0 + 48 * 512;

  v8f acc[4] = {V8F_ZERO, V8F_ZERO, V8F_ZERO, V8F_ZERO};
#pragma unroll 2
  for (int kc = 0; kc < 16; ++kc) {
    v16bf a = loadA_bf16(xrow, kc, half);           // shared by 4 WMMA chains
    acc[0] = wmma_bf16(a, loadB_bf16(w0, kc, half), acc[0]);
    acc[1] = wmma_bf16(a, loadB_bf16(w1, kc, half), acc[1]);
    acc[2] = wmma_bf16(a, loadB_bf16(w2, kc, half), acc[2]);
    acc[3] = wmma_bf16(a, loadB_bf16(w3, kc, half), acc[3]);
  }
  const int N = srow;
#pragma unroll
  for (int c = 0; c < 4; ++c) {
    const int col = cg * 64 + c * 16 + N;
#pragma unroll
    for (int r = 0; r < 8; ++r) {
      const int M = r + 8 * half;
      const int rowInB = rt * 16 + M;
      if (rowInB >= rowsPerB) continue;
      const float vres = acc[c][r];
      if constexpr (MODE == 0) {
        const int h = col >> 6, d = col & 63;
        const long long o = (((long long)(bb * 8 + h)) * 1024 + rowInB) * 64 + d;
        ((unsigned short*)out0)[o] = f2bf(vres + bias0[(h << 6) + d]);
        ((unsigned short*)out1)[o] = f2bf(vres + bias1[(h << 6) + d]);
      } else if constexpr (MODE == 1) {
        const int h = col >> 6, d = col & 63;
        ((unsigned short*)out0)[(((long long)(bb * 8 + h)) * 1024 + rowInB) * 64 + d] = f2bf(vres);
      } else if constexpr (MODE == 2) {
        const int h = col >> 6, d = col & 63;
        ((unsigned short*)out0)[(((long long)(bb * 8 + h)) * 64 + d) * 1024 + rowInB] = f2bf(vres);
      } else if constexpr (MODE == 3) {
        const int h = col >> 6, d = col & 63;
        ((unsigned short*)out0)[(((long long)(bb * 8 + h)) * 2048 + rowInB) * 64 + d] = f2bf(vres);
      } else {
        ((float*)out0)[(long long)rowInB * 512 + col] = vres;
      }
    }
  }
}

// ---------------- fused relative attention per (b,h,16-row query tile) -------
__global__ void attn_kernel(const unsigned short* __restrict__ qu,
                            const unsigned short* __restrict__ qv,
                            const unsigned short* __restrict__ kk,
                            const unsigned short* __restrict__ vt,
                            const unsigned short* __restrict__ pp,
                            unsigned short* __restrict__ ctx) {
  __shared__ __align__(16) unsigned short sc16[16 * 1024];  // score/attn bf16
  __shared__ float red[16 * 8];

  const int S = 1024;
  const int bh = blockIdx.x >> 6;
  const int s0 = (blockIdx.x & 63) * 16;
  const int b = bh >> 3, h = bh & 7;
  const int wave = threadIdx.x >> 5, lane = threadIdx.x & 31;
  const int srow = lane & 15, half = lane >> 4;

  // hoisted A-operands: (q+u) and (q+v) rows s0..s0+15, Dh=64 -> 2 K-chunks
  const unsigned short* quRow = qu + ((long long)bh * S + s0 + srow) * 64;
  const unsigned short* qvRow = qv + ((long long)bh * S + s0 + srow) * 64;
  v16bf aU0 = loadA_bf16(quRow, 0, half), aU1 = loadA_bf16(quRow, 1, half);
  v16bf aV0 = loadA_bf16(qvRow, 0, half), aV1 = loadA_bf16(qvRow, 1, half);

  const unsigned short* Kb = kk + (long long)bh * S * 64;
  const unsigned short* Pb = pp + (long long)bh * 2048 * 64;

  // Phase A: scores. shifted[s,t] = qu[s].k[t] + qv[s].p[S-1-s+t]
  for (int t0 = wave * 16; t0 < S; t0 += 64) {
    v8f cc = V8F_ZERO;
    cc = wmma_bf16(aU0, loadB_bf16(Kb + (t0 + srow) * 64, 0, half), cc);
    cc = wmma_bf16(aU1, loadB_bf16(Kb + (t0 + srow) * 64, 1, half), cc);

    const int j0 = 1008 - s0 + t0;  // window base: S-1-s0-15+t0, in [0, 2016]
    v8f p0 = V8F_ZERO, p1 = V8F_ZERO;
    p0 = wmma_bf16(aV0, loadB_bf16(Pb + (j0 + srow) * 64, 0, half), p0);
    p0 = wmma_bf16(aV1, loadB_bf16(Pb + (j0 + srow) * 64, 1, half), p0);
    p1 = wmma_bf16(aV0, loadB_bf16(Pb + (j0 + 16 + srow) * 64, 0, half), p1);
    p1 = wmma_bf16(aV1, loadB_bf16(Pb + (j0 + 16 + srow) * 64, 1, half), p1);

#pragma unroll
    for (int r = 0; r < 8; ++r) {
      const int M = r + 8 * half;
      const int w = 15 - M + srow;            // window col in [0,30]
      const int sl = (w & 15) + (half << 4);  // source lane, same half
      const float v0 = __shfl(p0[r], sl, 32);
      const float v1 = __shfl(p1[r], sl, 32);
      const float pos = (w >= 16) ? v1 : v0;
      sc16[M * 1024 + t0 + srow] = f2bf((cc[r] + pos) * 0.125f);
    }
  }
  __syncthreads();

  // Phase B: softmax stats over 16x1024 (row = tid&15, 8 chunks of 128 cols).
  // exp() stored unnormalized; 1/rowsum folded into phase-C epilogue.
  const int row = threadIdx.x & 15, chunk = threadIdx.x >> 4;
  const int base = row * 1024 + chunk * 128;
  float m = -1e30f;
  for (int c = 0; c < 128; c += 8) {
    BFrag f;
    f.q[0] = *(const U16B*)(sc16 + base + c);
#pragma unroll
    for (int j = 0; j < 8; ++j) m = fmaxf(m, bf2f(f.u[j]));
  }
  red[row * 8 + chunk] = m;
  __syncthreads();
  if (threadIdx.x < 16) {
    float mm = red[threadIdx.x * 8];
    for (int i = 1; i < 8; ++i) mm = fmaxf(mm, red[threadIdx.x * 8 + i]);
    red[threadIdx.x * 8] = mm;
  }
  __syncthreads();
  const float rm = red[row * 8];
  __syncthreads();
  float sum = 0.f;
  for (int c = 0; c < 128; c += 8) {
    BFrag f;
    f.q[0] = *(const U16B*)(sc16 + base + c);
#pragma unroll
    for (int j = 0; j < 8; ++j) {
      const float e = __expf(bf2f(f.u[j]) - rm);
      f.u[j] = f2bf(e);
      sum += e;
    }
    *(U16B*)(sc16 + base + c) = f.q[0];
  }
  red[row * 8 + chunk] = sum;
  __syncthreads();
  if (threadIdx.x < 16) {
    float t = 0.f;
    for (int i = 0; i < 8; ++i) t += red[threadIdx.x * 8 + i];
    red[threadIdx.x * 8] = 1.0f / t;
  }
  __syncthreads();

  // Phase C: context tile = attn[16,1024] @ V[1024,64]; wave -> 16 cols.
  // Two independent accumulation chains for WMMA ILP.
  const int d0 = wave * 16;
  v8f acc0 = V8F_ZERO, acc1 = V8F_ZERO;
  const unsigned short* Vb = vt + ((long long)bh * 64 + d0 + srow) * 1024;
  const unsigned short* Ar = sc16 + srow * 1024;
#pragma unroll 2
  for (int kc = 0; kc < 32; kc += 2) {
    acc0 = wmma_bf16(loadA_bf16(Ar, kc, half), loadB_bf16(Vb, kc, half), acc0);
    acc1 = wmma_bf16(loadA_bf16(Ar, kc + 1, half), loadB_bf16(Vb, kc + 1, half), acc1);
  }
#pragma unroll
  for (int r = 0; r < 8; ++r) {
    const int M = r + 8 * half;
    const float inv = red[M * 8];  // fold softmax normalization into output
    ctx[((long long)(b * S + s0 + M)) * 512 + h * 64 + d0 + srow] =
        f2bf((acc0[r] + acc1[r]) * inv);
  }
}

extern "C" void kernel_launch(void* const* d_in, const int* in_sizes, int n_in,
                              void* d_out, int out_size, void* d_ws, size_t ws_size,
                              hipStream_t stream) {
  const float* query = (const float*)d_in[0];
  const float* key   = (const float*)d_in[1];
  const float* value = (const float*)d_in[2];
  const float* pose  = (const float*)d_in[3];
  const float* Wq = (const float*)d_in[4];
  const float* Wk = (const float*)d_in[5];
  const float* Wv = (const float*)d_in[6];
  const float* Wp = (const float*)d_in[7];
  const float* u_bias = (const float*)d_in[8];
  const float* v_bias = (const float*)d_in[9];
  const float* Wo = (const float*)d_in[10];

  char* w = (char*)d_ws;
  size_t off = 0;
  auto carve = [&](size_t bytes) {
    void* p = w + off;
    off += (bytes + 255) & ~(size_t)255;
    return p;
  };
  const size_t WSZ = 512ull * 512 * 2;          // 512 KB
  const size_t QSZ = 8ull * 8 * 1024 * 64 * 2;  // 8 MB
  const size_t PSZ = 8ull * 8 * 2048 * 64 * 2;  // 16 MB
  const size_t XSZ = 8ull * 1024 * 512 * 2;     // 8 MB
  const size_t XPSZ = 8ull * 2047 * 512 * 2;    // ~16 MB
  unsigned short* wqT = (unsigned short*)carve(WSZ);
  unsigned short* wkT = (unsigned short*)carve(WSZ);
  unsigned short* wvT = (unsigned short*)carve(WSZ);
  unsigned short* wpT = (unsigned short*)carve(WSZ);
  unsigned short* woT = (unsigned short*)carve(WSZ);
  unsigned short* qu  = (unsigned short*)carve(QSZ);
  unsigned short* qv  = (unsigned short*)carve(QSZ);
  unsigned short* kk  = (unsigned short*)carve(QSZ);
  unsigned short* vt  = (unsigned short*)carve(QSZ);
  unsigned short* pp  = (unsigned short*)carve(PSZ);
  unsigned short* ctx = (unsigned short*)carve(XSZ);
  unsigned short* xq  = (unsigned short*)carve(XSZ);
  unsigned short* xk  = (unsigned short*)carve(XSZ);
  unsigned short* xv  = (unsigned short*)carve(XSZ);
  unsigned short* xp  = (unsigned short*)carve(XPSZ);

  // 1) weights transpose + bf16; zero p pad row; convert activations to bf16
  {
    const int total = 5 * 512 * 512 + 64 * 64;
    prep_kernel<<<(total + 255) / 256, 256, 0, stream>>>(Wq, Wk, Wv, Wp, Wo,
                                                         wqT, wkT, wvT, wpT, woT, pp);
    const int n8q = 8 * 1024 * 512 / 8;   // 524288
    const int n8p = 8 * 2047 * 512 / 8;   // 1048064
    cvt_bf16_kernel<<<(n8q + 255) / 256, 256, 0, stream>>>(query, xq, n8q);
    cvt_bf16_kernel<<<(n8q + 255) / 256, 256, 0, stream>>>(key, xk, n8q);
    cvt_bf16_kernel<<<(n8q + 255) / 256, 256, 0, stream>>>(value, xv, n8q);
    cvt_bf16_kernel<<<(n8p + 255) / 256, 256, 0, stream>>>(pose, xp, n8p);
  }
  // 2) projections: 16x64 strip per wave, 4 waves/block
  proj_gemm<0><<<8 * 64 * 8 / 4, TPB, 0, stream>>>(xq, wqT, qu, qv, u_bias, v_bias, 1024, 8, 64);
  proj_gemm<1><<<8 * 64 * 8 / 4, TPB, 0, stream>>>(xk, wkT, kk, nullptr, nullptr, nullptr, 1024, 8, 64);
  proj_gemm<2><<<8 * 64 * 8 / 4, TPB, 0, stream>>>(xv, wvT, vt, nullptr, nullptr, nullptr, 1024, 8, 64);
  proj_gemm<3><<<8 * 128 * 8 / 4, TPB, 0, stream>>>(xp, wpT, pp, nullptr, nullptr, nullptr, 2047, 8, 128);
  // 3) fused relative attention (B*H*S/16 = 4096 workgroups)
  attn_kernel<<<8 * 8 * 64, TPB, 0, stream>>>(qu, qv, kk, vt, pp, ctx);
  // 4) output projection -> fp32 d_out
  proj_gemm<4><<<512 * 8 / 4, TPB, 0, stream>>>(ctx, woT, d_out, nullptr, nullptr, nullptr, 8192, 1, 512);
}
